// SwitchPolicyValueNetwork_11793980194843
// MI455X (gfx1250) — matrix operand order repaired and checked by reading, not compiled, not run on hardware
//
#include <hip/hip_runtime.h>
#include <hip/hip_bf16.h>
#include <math.h>

// ---------------------------------------------------------------------------
// SwitchPolicyValueNetwork for MI455X (gfx1250, wave32, WMMA)
//   pre   : X  = tanh(obs @ pre_w^T + pre_b)            [4096,2048] bf16
//   expert: Y  = tanh(X_g @ expert_w[e]^T + expert_b[e])[4096,2048] bf16
//           (rows grouped by expert via permutation; only selected expert)
//   post  : out= tanh(Y @ post_w^T + post_b)            [4096,1024] f32
// Block tile 128x64, 8 waves, wave tile 32x32 (2x2 WMMA 16x16x32 bf16),
// double-buffered LDS, GLOBAL_LOAD_ASYNC_TO_LDS for bf16 A tiles.
// ---------------------------------------------------------------------------

#define B_ROWS  4096
#define IN_DIM  1024
#define H_DIM   2048
#define OUT_DIM 1024
#define NEXP    16

#define MT 128  // block tile M
#define NT 64   // block tile N
#define KT 32   // K step (matches 16x16x32 bf16 WMMA)

typedef __attribute__((ext_vector_type(16))) __bf16       v16bf;
typedef __attribute__((ext_vector_type(8)))  float        v8f;
typedef __attribute__((ext_vector_type(4)))  unsigned int u32x4;
typedef __attribute__((ext_vector_type(4)))  float        f32x4;
typedef __attribute__((ext_vector_type(4)))  int          v4i;

union FragBF { v16bf v; u32x4 q[2]; };
union Pack8  { u32x4 q; __bf16 h[8]; };
union Acc    { v8f  v; float  f[8]; };

// ---- optional CDNA5 async global->LDS path (guarded; sync fallback) -------
#if defined(__gfx1250__) && __has_builtin(__builtin_amdgcn_global_load_async_to_lds_b128)
#define HAVE_ASYNC_LDS 1
#define GPTR(p) ((__attribute__((address_space(1))) v4i*)(p))
#define LPTR(p) ((__attribute__((address_space(3))) v4i*)(p))
#if __has_builtin(__builtin_amdgcn_s_wait_asynccnt)
#define ASYNC_WAIT() __builtin_amdgcn_s_wait_asynccnt(0)
#else
#define ASYNC_WAIT() asm volatile("s_wait_asynccnt 0x0" ::: "memory")
#endif
#else
#define HAVE_ASYNC_LDS 0
#define ASYNC_WAIT() ((void)0)
#endif

__device__ __forceinline__ void cvt8_store(__bf16* dst, f32x4 a, f32x4 b) {
    Pack8 p;
    p.h[0] = (__bf16)a.x; p.h[1] = (__bf16)a.y; p.h[2] = (__bf16)a.z; p.h[3] = (__bf16)a.w;
    p.h[4] = (__bf16)b.x; p.h[5] = (__bf16)b.y; p.h[6] = (__bf16)b.z; p.h[7] = (__bf16)b.w;
    *(u32x4*)dst = p.q;
}

// A fragment (16x32 bf16): lane&15 = M row; lane-half selects the ISA's
// K 0-7/16-23 vs 8-15/24-31 interleave. B fragment (32x16): lane&15 = N col;
// lane-half selects the same K 0-7/16-23 vs 8-15/24-31 interleave.
__device__ __forceinline__ void load_a_frag(const __bf16* As, int rowBase,
                                            int l15, int hi, FragBF& a) {
    const __bf16* ap = As + (rowBase + l15) * KT;
    a.q[0] = *(const u32x4*)(ap + (hi << 3));
    a.q[1] = *(const u32x4*)(ap + 16 + (hi << 3));
}
__device__ __forceinline__ void load_b_frag(const __bf16* Bs, int nBase,
                                            int l15, int hi, FragBF& b) {
    const __bf16* bp = Bs + (nBase + l15) * KT + (hi << 3);
    b.q[0] = *(const u32x4*)(bp);
    b.q[1] = *(const u32x4*)(bp + 16);
}

// One K-step for one wave: 32x32 output patch = 2x2 WMMA 16x16x32.
__device__ __forceinline__ void mma_step(const __bf16* As, const __bf16* Bs,
                                         int waveRow, int waveCol, int lane,
                                         Acc acc[2][2]) {
    const int l15 = lane & 15, hi = lane >> 4;
    FragBF a0, a1, b0, b1;
    load_a_frag(As, (waveRow << 5),      l15, hi, a0);
    load_a_frag(As, (waveRow << 5) + 16, l15, hi, a1);
    load_b_frag(Bs, (waveCol << 5),      l15, hi, b0);
    load_b_frag(Bs, (waveCol << 5) + 16, l15, hi, b1);
    acc[0][0].v = __builtin_amdgcn_wmma_f32_16x16x32_bf16(false, a0.v, false, b0.v,
                                                          (short)0, acc[0][0].v, false, false);
    acc[0][1].v = __builtin_amdgcn_wmma_f32_16x16x32_bf16(false, a0.v, false, b1.v,
                                                          (short)0, acc[0][1].v, false, false);
    acc[1][0].v = __builtin_amdgcn_wmma_f32_16x16x32_bf16(false, a1.v, false, b0.v,
                                                          (short)0, acc[1][0].v, false, false);
    acc[1][1].v = __builtin_amdgcn_wmma_f32_16x16x32_bf16(false, a1.v, false, b1.v,
                                                          (short)0, acc[1][1].v, false, false);
}

// ---------------------------------------------------------------------------
// Dense GEMM: C[M,N] = tanh(A[M,K] @ W[N,K]^T + bias[N])
// ---------------------------------------------------------------------------
template<bool A_BF16, bool C_BF16>
__global__ __launch_bounds__(256)
void gemm_bias_tanh(const void* __restrict__ Ap, const float* __restrict__ W,
                    const float* __restrict__ bias, void* __restrict__ Cp,
                    int N, int K)
{
    __shared__ __bf16 As[2][MT * KT];   // 2 x 8 KB
    __shared__ __bf16 Bs[2][NT * KT];   // 2 x 4 KB

    const int tid  = threadIdx.x;
    const int lane = tid & 31;
    const int wave = tid >> 5;
    const int waveRow = wave & 3;    // 4 M-subtiles of 32
    const int waveCol = wave >> 2;   // 2 N-subtiles of 32
    const int m0 = blockIdx.y * MT;
    const int n0 = blockIdx.x * NT;

    const int arow = tid >> 1;          // 0..127, 16 bf16 per thread
    const int acol = (tid & 1) << 4;    // 0 or 16
    const int brow = tid >> 2;          // 0..63, 8 f32 per thread
    const int bcol = (tid & 3) << 3;    // 0,8,16,24

    Acc acc[2][2];
#pragma unroll
    for (int i = 0; i < 2; ++i)
#pragma unroll
        for (int j = 0; j < 2; ++j)
#pragma unroll
            for (int r = 0; r < 8; ++r) acc[i][j].f[r] = 0.f;

    auto stage = [&](int b, int k0) {
        if (A_BF16) {
            const __bf16* src = (const __bf16*)Ap + (size_t)(m0 + arow) * K + k0 + acol;
            __bf16* dst = &As[b][arow * KT + acol];
#if HAVE_ASYNC_LDS
            __builtin_amdgcn_global_load_async_to_lds_b128(GPTR(src), LPTR(dst), 0, 0);
            __builtin_amdgcn_global_load_async_to_lds_b128(GPTR(src), LPTR(dst), 16, 0);
#else
            ((u32x4*)dst)[0] = ((const u32x4*)src)[0];
            ((u32x4*)dst)[1] = ((const u32x4*)src)[1];
#endif
        } else {
            const f32x4* p = (const f32x4*)((const float*)Ap + (size_t)(m0 + arow) * K + k0 + acol);
            __bf16* dst = &As[b][arow * KT + acol];
            cvt8_store(dst,     p[0], p[1]);
            cvt8_store(dst + 8, p[2], p[3]);
        }
        const f32x4* q = (const f32x4*)(W + (size_t)(n0 + brow) * K + k0 + bcol);
        cvt8_store(&Bs[b][brow * KT + bcol], q[0], q[1]);
    };

    const int NKT = K / KT;
    stage(0, 0);
    for (int kt = 0; kt < NKT; ++kt) {
        ASYNC_WAIT();
        __syncthreads();
        if (kt + 1 < NKT) stage((kt + 1) & 1, (kt + 1) * KT);
        mma_step(As[kt & 1], Bs[kt & 1], waveRow, waveCol, lane, acc);
    }

    const int l15 = lane & 15, hi = lane >> 4;
#pragma unroll
    for (int ms = 0; ms < 2; ++ms)
#pragma unroll
        for (int ns = 0; ns < 2; ++ns) {
            const int n  = n0 + (waveCol << 5) + (ns << 4) + l15;
            const float bn = bias[n];
#pragma unroll
            for (int r = 0; r < 8; ++r) {
                const int m = m0 + (waveRow << 5) + (ms << 4) + (hi << 3) + r;
                const float v = tanhf(acc[ms][ns].f[r] + bn);
                if (C_BF16) ((__bf16*)Cp)[(size_t)m * N + n] = (__bf16)v;
                else        ((float*) Cp)[(size_t)m * N + n] = v;
            }
        }
}

// ---------------------------------------------------------------------------
// meta layout (ints), at d_ws + 32MB:
//  [0..15] counts  [16..31] offsets  [32..47] cursor  [48] numTiles
//  [64..383]   tile table: int4 {expert, rowStart, rowCount, 0} (<=48 used)
//  [384..4479] perm: row indices grouped by expert
// ---------------------------------------------------------------------------
__global__ void k_zero_meta(int* meta) {
    if (threadIdx.x < 64) meta[threadIdx.x] = 0;
}

__global__ void k_hist(const int* __restrict__ idx, int* __restrict__ meta) {
    const int i = blockIdx.x * 256 + threadIdx.x;
    if (i < B_ROWS) atomicAdd(&meta[idx[i] & (NEXP - 1)], 1);
}

__global__ void k_build(int* meta) {
    if (threadIdx.x == 0 && blockIdx.x == 0) {
        int off = 0, nt = 0;
        int4* tab = (int4*)(meta + 64);
        for (int e = 0; e < NEXP; ++e) {
            const int c = meta[e];
            meta[16 + e] = off;
            meta[32 + e] = off;
            const int tiles = (c + MT - 1) / MT;
            for (int t = 0; t < tiles; ++t) {
                int4 ent;
                ent.x = e;
                ent.y = off + t * MT;
                ent.z = (c - t * MT) > MT ? MT : (c - t * MT);
                ent.w = 0;
                tab[nt++] = ent;
            }
            off += c;
        }
        meta[48] = nt;
    }
}

__global__ void k_scatter(const int* __restrict__ idx, int* __restrict__ meta) {
    const int i = blockIdx.x * 256 + threadIdx.x;
    if (i < B_ROWS) {
        const int e = idx[i] & (NEXP - 1);
        const int p = atomicAdd(&meta[32 + e], 1);
        meta[384 + p] = i;
    }
}

// ---------------------------------------------------------------------------
// Expert GEMM (gathered rows, scattered stores):
//   Y[b,:] = tanh(X[b,:] @ expert_w[e]^T + expert_b[e]),  e = switch_idx[b]
// ---------------------------------------------------------------------------
__global__ __launch_bounds__(256)
void expert_gemm(const __bf16* __restrict__ X, const float* __restrict__ EW,
                 const float* __restrict__ EB, __bf16* __restrict__ Y,
                 const int* __restrict__ meta)
{
    if ((int)blockIdx.y >= meta[48]) return;
    const int4 ent = ((const int4*)(meta + 64))[blockIdx.y];
    const int e = ent.x, rowStart = ent.y, cnt = ent.z;
    const int* perm = meta + 384;
    const float* W    = EW + (size_t)e * H_DIM * H_DIM;
    const float* bias = EB + (size_t)e * H_DIM;

    __shared__ __bf16 As[2][MT * KT];
    __shared__ __bf16 Bs[2][NT * KT];

    const int tid  = threadIdx.x;
    const int lane = tid & 31;
    const int wave = tid >> 5;
    const int waveRow = wave & 3;
    const int waveCol = wave >> 2;
    const int n0 = blockIdx.x * NT;

    const int arow = tid >> 1;
    const int acol = (tid & 1) << 4;
    const int brow = tid >> 2;
    const int bcol = (tid & 3) << 3;

    const int rl   = arow < cnt ? arow : (cnt - 1);   // clamp tail rows
    const int grow = perm[rowStart + rl];             // gathered batch row

    Acc acc[2][2];
#pragma unroll
    for (int i = 0; i < 2; ++i)
#pragma unroll
        for (int j = 0; j < 2; ++j)
#pragma unroll
            for (int r = 0; r < 8; ++r) acc[i][j].f[r] = 0.f;

    auto stage = [&](int b, int k0) {
        const __bf16* src = X + (size_t)grow * H_DIM + k0 + acol;
        __bf16* dst = &As[b][arow * KT + acol];
#if HAVE_ASYNC_LDS
        __builtin_amdgcn_global_load_async_to_lds_b128(GPTR(src), LPTR(dst), 0, 0);
        __builtin_amdgcn_global_load_async_to_lds_b128(GPTR(src), LPTR(dst), 16, 0);
#else
        ((u32x4*)dst)[0] = ((const u32x4*)src)[0];
        ((u32x4*)dst)[1] = ((const u32x4*)src)[1];
#endif
        const f32x4* q = (const f32x4*)(W + (size_t)(n0 + brow) * H_DIM + k0 + bcol);
        cvt8_store(&Bs[b][brow * KT + bcol], q[0], q[1]);
    };

    const int NKT = H_DIM / KT;
    stage(0, 0);
    for (int kt = 0; kt < NKT; ++kt) {
        ASYNC_WAIT();
        __syncthreads();
        if (kt + 1 < NKT) stage((kt + 1) & 1, (kt + 1) * KT);
        mma_step(As[kt & 1], Bs[kt & 1], waveRow, waveCol, lane, acc);
    }

    const int l15 = lane & 15, hi = lane >> 4;
#pragma unroll
    for (int ms = 0; ms < 2; ++ms)
#pragma unroll
        for (int ns = 0; ns < 2; ++ns) {
            const int n  = n0 + (waveCol << 5) + (ns << 4) + l15;
            const float bn = bias[n];
#pragma unroll
            for (int r = 0; r < 8; ++r) {
                const int ml = (waveRow << 5) + (ms << 4) + (hi << 3) + r;
                if (ml < cnt) {
                    const int brow_g = perm[rowStart + ml];
                    Y[(size_t)brow_g * H_DIM + n] = (__bf16)tanhf(acc[ms][ns].f[r] + bn);
                }
            }
        }
}

// ---------------------------------------------------------------------------
extern "C" void kernel_launch(void* const* d_in, const int* in_sizes, int n_in,
                              void* d_out, int out_size, void* d_ws, size_t ws_size,
                              hipStream_t stream) {
    const float* obs    = (const float*)d_in[0];
    const int*   swidx  = (const int*)  d_in[1];
    const float* pre_w  = (const float*)d_in[2];
    const float* pre_b  = (const float*)d_in[3];
    const float* ex_w   = (const float*)d_in[4];
    const float* ex_b   = (const float*)d_in[5];
    const float* post_w = (const float*)d_in[6];
    const float* post_b = (const float*)d_in[7];
    float* out = (float*)d_out;

    char* ws = (char*)d_ws;
    __bf16* X    = (__bf16*)ws;                                   // 16 MB
    __bf16* Y    = (__bf16*)(ws + (size_t)B_ROWS * H_DIM * 2);    // 16 MB
    int*    meta = (int*)   (ws + (size_t)B_ROWS * H_DIM * 4);    // ~18 KB

    // Build expert grouping metadata (overlaps with pre-GEMM on the stream).
    k_zero_meta<<<1, 64, 0, stream>>>(meta);
    k_hist<<<B_ROWS / 256, 256, 0, stream>>>(swidx, meta);

    // pre: X = tanh(obs @ pre_w^T + pre_b)   [f32 A -> bf16 C]
    gemm_bias_tanh<false, true>
        <<<dim3(H_DIM / NT, B_ROWS / MT), 256, 0, stream>>>(
            obs, pre_w, pre_b, X, H_DIM, IN_DIM);

    k_build<<<1, 1, 0, stream>>>(meta);
    k_scatter<<<B_ROWS / 256, 256, 0, stream>>>(swidx, meta);

    // expert: Y = tanh(X_gathered @ expert_w[e]^T + expert_b[e])
    expert_gemm<<<dim3(H_DIM / NT, B_ROWS / MT + NEXP), 256, 0, stream>>>(
        X, ex_w, ex_b, Y, meta);

    // post: out = tanh(Y @ post_w^T + post_b)  [bf16 A -> f32 C]
    gemm_bias_tanh<true, false>
        <<<dim3(OUT_DIM / NT, B_ROWS / MT), 256, 0, stream>>>(
            Y, post_w, post_b, out, OUT_DIM, H_DIM);
}
